// GATV2_52656299049252
// MI455X (gfx1250) — compile-verified, matrix operand order checked by reading, not run on hardware
//
#include <hip/hip_runtime.h>
#include <hip/hip_bf16.h>
#include <math.h>

// ---------------------------------------------------------------------------
// Types for CDNA5 WMMA (wave32)
// ---------------------------------------------------------------------------
typedef _Float16 half8  __attribute__((ext_vector_type(8)));
typedef _Float16 v16h_t __attribute__((ext_vector_type(16)));
typedef float    v8f_t  __attribute__((ext_vector_type(8)));

#define FFW    128
#define NHEADS 8
#define HCH    16
#define NGRAPH 256

// ---------------------------------------------------------------------------
// Small device helpers
// ---------------------------------------------------------------------------
__device__ __forceinline__ float gelu_exact(float x) {
    return 0.5f * x * (1.0f + erff(x * 0.70710678118654752440f));
}
// monotone float<->uint mapping for atomicMax on floats
__device__ __forceinline__ unsigned encF(float f) {
    unsigned u = __float_as_uint(f);
    return (u & 0x80000000u) ? ~u : (u | 0x80000000u);
}
__device__ __forceinline__ float decF(unsigned u) {
    u = (u & 0x80000000u) ? (u & 0x7FFFFFFFu) : ~u;
    return __uint_as_float(u);
}
__device__ __forceinline__ void edge_ends(const int* __restrict__ ei, int E,
                                          int e, int& s, int& d) {
    if (e < E) { s = ei[e]; d = ei[E + e]; }
    else       { s = e - E; d = e - E; }           // self loops appended
}

// ---------------------------------------------------------------------------
// Generic fills / conversions
// ---------------------------------------------------------------------------
__global__ void fill_f32_k(float* p, float v, long long n) {
    long long i = (long long)blockIdx.x * blockDim.x + threadIdx.x;
    if (i < n) p[i] = v;
}
__global__ void fill_u32_k(unsigned* p, unsigned v, long long n) {
    long long i = (long long)blockIdx.x * blockDim.x + threadIdx.x;
    if (i < n) p[i] = v;
}
__global__ void cvt16_k(const float* __restrict__ s, _Float16* __restrict__ d,
                        long long n) {
    long long i = (long long)blockIdx.x * blockDim.x + threadIdx.x;
    if (i < n) d[i] = (_Float16)s[i];
}
// W [K x Nn] f32 row-major  ->  Wt [Nn x K] f16 row-major (transposed)
__global__ void tcvt_k(const float* __restrict__ W, _Float16* __restrict__ Wt,
                       int K, int Nn) {
    int t = blockIdx.x * blockDim.x + threadIdx.x;
    if (t >= K * Nn) return;
    int n = t / K, k = t % K;
    Wt[t] = (_Float16)W[(size_t)k * Nn + n];
}

// ---------------------------------------------------------------------------
// WMMA GEMM: C[M,Nn] = A[M,K](f16) * Bt[Nn,K]^T(f16) (+ epilogue)
//   EPI 0: C = acc + bias[col]
//   EPI 1: C = acc
//   EPI 2: C += ls[0] * (acc + bias[col])      (fused residual)
// One wave computes a 16x16 tile; 8 waves/block cover column tiles.
// Requires M %16==0, Nn %16==0, K %32==0 (true for this model).
// ---------------------------------------------------------------------------
template <int EPI>
__global__ __launch_bounds__(256) void gemm_wmma(
    const _Float16* __restrict__ A, const _Float16* __restrict__ Bt,
    const float* __restrict__ bias, const float* __restrict__ ls,
    float* __restrict__ C, int K, int Nn) {
    const int wave = threadIdx.x >> 5;
    const int lane = threadIdx.x & 31;
    const int hi   = lane >> 4;         // 0/1 : which K-half group this lane holds
    const int l15  = lane & 15;
    const int kb   = hi * 8;
    const size_t rowA = (size_t)blockIdx.x * 16 + l15;
    const int ntiles = Nn >> 4;
    const _Float16* ap = A + rowA * (size_t)K + kb;

    for (int tn = wave; tn < ntiles; tn += 8) {
        const int col = tn * 16 + l15;
        const _Float16* bp = Bt + (size_t)col * K + kb;
        v8f_t acc = {};
        for (int kk = 0; kk < K; kk += 32) {
            half8 a0 = *(const half8*)(ap + kk);
            half8 a1 = *(const half8*)(ap + kk + 16);
            half8 b0 = *(const half8*)(bp + kk);
            half8 b1 = *(const half8*)(bp + kk + 16);
            __builtin_prefetch(bp + kk + 32, 0, 0);   // global_prefetch_b8
            v16h_t av = __builtin_shufflevector(a0, a1, 0,1,2,3,4,5,6,7,
                                                        8,9,10,11,12,13,14,15);
            v16h_t bv = __builtin_shufflevector(b0, b1, 0,1,2,3,4,5,6,7,
                                                        8,9,10,11,12,13,14,15);
            acc = __builtin_amdgcn_wmma_f32_16x16x32_f16(
                false, av, false, bv, (short)0, acc, false, false);
        }
        const size_t row0 = (size_t)blockIdx.x * 16 + hi * 8;
        float* cp = C + row0 * Nn + col;
        float bcol;
        if constexpr (EPI == 1) bcol = 0.0f; else bcol = bias[col];
        if constexpr (EPI == 2) {
            const float l = ls[0];
#pragma unroll
            for (int r = 0; r < 8; ++r) cp[(size_t)r * Nn] += l * (acc[r] + bcol);
        } else {
#pragma unroll
            for (int r = 0; r < 8; ++r) cp[(size_t)r * Nn] = acc[r] + bcol;
        }
    }
}

// ---------------------------------------------------------------------------
// GAT attention pieces
// ---------------------------------------------------------------------------
// per-node per-head projections onto att_src / att_dst
__global__ void attn_scores_k(const float* __restrict__ hfeat,
                              const float* __restrict__ asrc,
                              const float* __restrict__ adst,
                              float* __restrict__ ss, float* __restrict__ sd,
                              int Nn) {
    int t = blockIdx.x * blockDim.x + threadIdx.x;
    if (t >= Nn * NHEADS) return;
    int n = t >> 3, h = t & 7;
    const float* hp = hfeat + (size_t)n * FFW + h * HCH;
    float a = 0.f, b = 0.f;
#pragma unroll
    for (int c = 0; c < HCH; ++c) {
        float v = hp[c];
        a += v * asrc[h * HCH + c];
        b += v * adst[h * HCH + c];
    }
    ss[t] = a; sd[t] = b;
}
__global__ void edge_max_k(const int* __restrict__ ei,
                           const float* __restrict__ ss,
                           const float* __restrict__ sd,
                           unsigned* __restrict__ m8, int E, int Etot) {
    long long t = (long long)blockIdx.x * blockDim.x + threadIdx.x;
    if (t >= (long long)Etot * NHEADS) return;
    int e = (int)(t >> 3), h = (int)(t & 7);
    int s, d; edge_ends(ei, E, e, s, d);
    float a = ss[s * 8 + h] + sd[d * 8 + h];
    a = a > 0.f ? a : 0.2f * a;                 // leaky_relu(0.2)
    atomicMax(&m8[d * 8 + h], encF(a));
}
__global__ void decode_max_k(unsigned* __restrict__ m, long long n) {
    long long i = (long long)blockIdx.x * blockDim.x + threadIdx.x;
    if (i >= n) return;
    float v = decF(m[i]);
    if (!isfinite(v)) v = 0.f;                  // ref: where(isfinite(m), m, 0)
    ((float*)m)[i] = v;
}
__global__ void edge_sum_k(const int* __restrict__ ei,
                           const float* __restrict__ ss,
                           const float* __restrict__ sd,
                           const float* __restrict__ mf,
                           float* __restrict__ s8, int E, int Etot) {
    long long t = (long long)blockIdx.x * blockDim.x + threadIdx.x;
    if (t >= (long long)Etot * NHEADS) return;
    int e = (int)(t >> 3), h = (int)(t & 7);
    int s, d; edge_ends(ei, E, e, s, d);
    float a = ss[s * 8 + h] + sd[d * 8 + h];
    a = a > 0.f ? a : 0.2f * a;
    atomicAdd(&s8[d * 8 + h], expf(a - mf[d * 8 + h]));
}
__global__ void edge_scatter_k(const int* __restrict__ ei,
                               const float* __restrict__ ss,
                               const float* __restrict__ sd,
                               const float* __restrict__ mf,
                               const float* __restrict__ s8,
                               const float* __restrict__ hfeat,
                               float* __restrict__ att, int E, int Etot) {
    long long t = (long long)blockIdx.x * blockDim.x + threadIdx.x;
    if (t >= (long long)Etot * FFW) return;
    int e = (int)(t >> 7), c = (int)(t & 127), h = c >> 4;
    int s, d; edge_ends(ei, E, e, s, d);
    float a = ss[s * 8 + h] + sd[d * 8 + h];
    a = a > 0.f ? a : 0.2f * a;
    float alpha = expf(a - mf[d * 8 + h]) / (s8[d * 8 + h] + 1e-16f);
    atomicAdd(&att[(size_t)d * FFW + c], hfeat[(size_t)s * FFW + c] * alpha);
}

// ---------------------------------------------------------------------------
// Fused elementwise stages
// ---------------------------------------------------------------------------
// att = gelu(batchnorm(att + conv_bias))
__global__ void bn_gelu_k(float* __restrict__ att, const float* __restrict__ cb,
                          const float* __restrict__ g, const float* __restrict__ b,
                          const float* __restrict__ m, const float* __restrict__ v,
                          long long total) {
    long long i = (long long)blockIdx.x * blockDim.x + threadIdx.x;
    if (i >= total) return;
    int c = (int)(i & 127);
    float x = att[i] + cb[c];
    x = (x - m[c]) * rsqrtf(v[c] + 1e-5f) * g[c] + b[c];
    att[i] = gelu_exact(x);
}
template <bool FIRST>
__global__ void resid_k(float* __restrict__ h, const float* __restrict__ ident,
                        const float* __restrict__ att, const float* __restrict__ ls,
                        long long total) {
    long long i = (long long)blockIdx.x * blockDim.x + threadIdx.x;
    if (i >= total) return;
    if constexpr (FIRST) h[i] = ident[i] + ls[0] * att[i];
    else                 h[i] += ls[0] * att[i];
}
// row-wise layernorm + gelu (blockDim.x == D, D in {128,256})
__global__ void ln_gelu_k(float* __restrict__ z, const float* __restrict__ g,
                          const float* __restrict__ b, int D) {
    __shared__ float sm[256];
    int c = threadIdx.x;
    float* zp = z + (size_t)blockIdx.x * D;
    float x = zp[c];
    sm[c] = x; __syncthreads();
    for (int s = D >> 1; s > 0; s >>= 1) {
        if (c < s) sm[c] += sm[c + s];
        __syncthreads();
    }
    float mu = sm[0] / (float)D; __syncthreads();
    float dx = x - mu;
    sm[c] = dx * dx; __syncthreads();
    for (int s = D >> 1; s > 0; s >>= 1) {
        if (c < s) sm[c] += sm[c + s];
        __syncthreads();
    }
    float var = sm[0] / (float)D;
    zp[c] = gelu_exact(dx * rsqrtf(var + 1e-5f) * g[c] + b[c]);
}

// ---------------------------------------------------------------------------
// Global pooling over graphs + heads
// ---------------------------------------------------------------------------
__global__ void pool_accum_k(const float* __restrict__ h,
                             const int* __restrict__ batch,
                             float* __restrict__ psum, unsigned* __restrict__ pmax,
                             float* __restrict__ cnt, int Nn) {
    long long t = (long long)blockIdx.x * blockDim.x + threadIdx.x;
    if (t >= (long long)Nn * FFW) return;
    int n = (int)(t >> 7), c = (int)(t & 127);
    int g = batch[n];
    float v = h[t];
    atomicAdd(&psum[g * FFW + c], v);
    atomicMax(&pmax[g * FFW + c], encF(v));
    if (c == 0) atomicAdd(&cnt[g], 1.0f);
}
__global__ void pool_final_k(const float* __restrict__ psum,
                             const unsigned* __restrict__ pmax,
                             const float* __restrict__ cnt,
                             float* __restrict__ pooled) {
    int t = blockIdx.x * blockDim.x + threadIdx.x;
    if (t >= NGRAPH * FFW) return;
    int g = t >> 7, c = t & 127;
    float s = psum[t];
    float mx = decF(pmax[t]);
    if (!isfinite(mx)) mx = 0.f;
    pooled[(size_t)g * 384 + c]       = s / fmaxf(cnt[g], 1.f);  // mean
    pooled[(size_t)g * 384 + 128 + c] = mx;                      // max
    pooled[(size_t)g * 384 + 256 + c] = s;                       // sum
}
// final tiny classifier GEMM (nc = 2 or 86, not WMMA-shaped)
__global__ void head_out_k(const float* __restrict__ z, const float* __restrict__ w2,
                           const float* __restrict__ b2, float* __restrict__ out,
                           int nc) {
    int t = blockIdx.x * blockDim.x + threadIdx.x;
    if (t >= NGRAPH * nc) return;
    int g = t / nc, j = t % nc;
    const float* zp = z + (size_t)g * FFW;
    float acc = b2[j];
    for (int k = 0; k < FFW; ++k) acc += zp[k] * w2[k * nc + j];
    out[t] = acc;
}

// ---------------------------------------------------------------------------
// Host orchestration
// ---------------------------------------------------------------------------
extern "C" void kernel_launch(void* const* d_in, const int* in_sizes, int n_in,
                              void* d_out, int out_size, void* d_ws, size_t ws_size,
                              hipStream_t stream) {
    (void)n_in; (void)out_size; (void)ws_size;
    const int Nn   = in_sizes[0] / 128;   // 50000
    const int E    = in_sizes[1] / 2;     // 600000
    const int Etot = E + Nn;              // + self loops
    const long long NF  = (long long)Nn * 128;
    const long long NF2 = (long long)Nn * 256;

    // ---- input pointers (setup_inputs dict order, recursive insertion order)
    const float* x    = (const float*)d_in[0];
    const int*   ei   = (const int*)d_in[1];
    const int*   batch= (const int*)d_in[2];
    const float* proj_w = (const float*)d_in[3];
    const float* proj_b = (const float*)d_in[4];
    struct ConvP { const float *w, *asrc, *adst, *b; } conv[3];
    for (int i = 0; i < 3; ++i) {
        int base = 5 + 4 * i;
        conv[i] = { (const float*)d_in[base], (const float*)d_in[base + 1],
                    (const float*)d_in[base + 2], (const float*)d_in[base + 3] };
    }
    struct BnP { const float *g, *b, *m, *v; } bn[3];
    for (int i = 0; i < 3; ++i) {
        int base = 17 + 4 * i;
        bn[i] = { (const float*)d_in[base], (const float*)d_in[base + 1],
                  (const float*)d_in[base + 2], (const float*)d_in[base + 3] };
    }
    struct FfP { const float *w1, *b1, *lng, *lnb, *w2, *b2; } ff[2];
    for (int i = 0; i < 2; ++i) {
        int base = 29 + 6 * i;
        ff[i] = { (const float*)d_in[base], (const float*)d_in[base + 1],
                  (const float*)d_in[base + 2], (const float*)d_in[base + 3],
                  (const float*)d_in[base + 4], (const float*)d_in[base + 5] };
    }
    const float* ls[3] = { (const float*)d_in[41], (const float*)d_in[42],
                           (const float*)d_in[43] };
    struct HeadP { const float *w1, *b1, *lng, *lnb, *w2, *b2; } hd[2];
    for (int i = 0; i < 2; ++i) {
        int base = 44 + 6 * i;
        hd[i] = { (const float*)d_in[base], (const float*)d_in[base + 1],
                  (const float*)d_in[base + 2], (const float*)d_in[base + 3],
                  (const float*)d_in[base + 4], (const float*)d_in[base + 5] };
    }
    const int ncs[2] = { 2, 86 };
    float* out_f = (float*)d_out;
    float* out_ptr[2] = { out_f, out_f + NGRAPH * ncs[0] };

    // ---- workspace layout (256B aligned slices)
    char* p = (char*)d_ws;
    auto take = [&](size_t bytes) -> char* {
        char* r = p; p += (bytes + 255) & ~(size_t)255; return r;
    };
    float*    H    = (float*)take(NF * 4);
    float*    T1   = (float*)take(NF * 4);          // hfeat = x @ conv.w
    float*    ATT  = (float*)take(NF * 4);          // scatter accumulator
    float*    Z    = (float*)take(NF2 * 4);         // ff hidden
    char*     IDZ  = take(NF * 4);                  // identity; later reused as Z16
    float*    IDENT = (float*)IDZ;
    _Float16* Z16  = (_Float16*)IDZ;                // N*256*2B == N*128*4B
    _Float16* H16  = (_Float16*)take(NF * 2);
    float*    SS   = (float*)take((size_t)Nn * 8 * 4);
    float*    SD   = (float*)take((size_t)Nn * 8 * 4);
    unsigned* M8   = (unsigned*)take((size_t)Nn * 8 * 4); // enc max -> f32 in place
    float*    S8   = (float*)take((size_t)Nn * 8 * 4);
    _Float16* projWT = (_Float16*)take(128 * 128 * 2);
    _Float16* convWT[3];
    for (int i = 0; i < 3; ++i) convWT[i] = (_Float16*)take(128 * 128 * 2);
    _Float16* ffW1T[2]; _Float16* ffW2T[2];
    for (int i = 0; i < 2; ++i) {
        ffW1T[i] = (_Float16*)take(256 * 128 * 2);  // [256,128]
        ffW2T[i] = (_Float16*)take(128 * 256 * 2);  // [128,256]
    }
    _Float16* hdW1T[2];
    for (int i = 0; i < 2; ++i) hdW1T[i] = (_Float16*)take(128 * 384 * 2);
    float*    psum  = (float*)take(NGRAPH * 128 * 4);
    unsigned* pmax  = (unsigned*)take(NGRAPH * 128 * 4);
    float*    cnt   = (float*)take(NGRAPH * 4);
    float*    pooled = (float*)take(NGRAPH * 384 * 4);
    _Float16* pooled16 = (_Float16*)take(NGRAPH * 384 * 2);
    float*    ZH    = (float*)take(NGRAPH * 128 * 4);

    auto g1 = [](long long n) { return dim3((unsigned)((n + 255) / 256)); };
    const unsigned ENC_NEG_INF = 0x007FFFFFu;   // encF(-inf)
    const int MBLK = Nn / 16;                   // 50000/16 = 3125

    // ---- weight prep: transpose + f32->f16
    tcvt_k<<<g1(128 * 128), 256, 0, stream>>>(proj_w, projWT, 128, 128);
    for (int i = 0; i < 3; ++i)
        tcvt_k<<<g1(128 * 128), 256, 0, stream>>>(conv[i].w, convWT[i], 128, 128);
    for (int i = 0; i < 2; ++i) {
        tcvt_k<<<g1(128 * 256), 256, 0, stream>>>(ff[i].w1, ffW1T[i], 128, 256);
        tcvt_k<<<g1(256 * 128), 256, 0, stream>>>(ff[i].w2, ffW2T[i], 256, 128);
    }
    for (int i = 0; i < 2; ++i)
        tcvt_k<<<g1(384 * 128), 256, 0, stream>>>(hd[i].w1, hdW1T[i], 384, 128);

    // ---- x -> f16; identity = x @ proj_w + proj_b
    cvt16_k<<<g1(NF), 256, 0, stream>>>(x, H16, NF);
    gemm_wmma<0><<<MBLK, 256, 0, stream>>>(H16, projWT, proj_b, nullptr, IDENT, 128, 128);

    // ---- one GAT + (optional) FF layer
    auto run_layer = [&](int li, bool has_ff, int ffi) {
        // hfeat = h @ conv.w (A operand currently in H16; x for layer 0)
        gemm_wmma<1><<<MBLK, 256, 0, stream>>>(H16, convWT[li], nullptr, nullptr, T1, 128, 128);
        attn_scores_k<<<g1((long long)Nn * 8), 256, 0, stream>>>(
            T1, conv[li].asrc, conv[li].adst, SS, SD, Nn);
        fill_u32_k<<<g1((long long)Nn * 8), 256, 0, stream>>>(M8, ENC_NEG_INF, (long long)Nn * 8);
        fill_f32_k<<<g1((long long)Nn * 8), 256, 0, stream>>>(S8, 0.f, (long long)Nn * 8);
        fill_f32_k<<<g1(NF), 256, 0, stream>>>(ATT, 0.f, NF);
        edge_max_k<<<g1((long long)Etot * 8), 256, 0, stream>>>(ei, SS, SD, M8, E, Etot);
        decode_max_k<<<g1((long long)Nn * 8), 256, 0, stream>>>(M8, (long long)Nn * 8);
        edge_sum_k<<<g1((long long)Etot * 8), 256, 0, stream>>>(ei, SS, SD, (float*)M8, S8, E, Etot);
        edge_scatter_k<<<g1((long long)Etot * 128), 256, 0, stream>>>(
            ei, SS, SD, (float*)M8, S8, T1, ATT, E, Etot);
        bn_gelu_k<<<g1(NF), 256, 0, stream>>>(ATT, conv[li].b, bn[li].g, bn[li].b,
                                              bn[li].m, bn[li].v, NF);
        if (li == 0)
            resid_k<true><<<g1(NF), 256, 0, stream>>>(H, IDENT, ATT, ls[li], NF);
        else
            resid_k<false><<<g1(NF), 256, 0, stream>>>(H, nullptr, ATT, ls[li], NF);
        if (has_ff) {
            cvt16_k<<<g1(NF), 256, 0, stream>>>(H, H16, NF);
            gemm_wmma<0><<<MBLK, 256, 0, stream>>>(H16, ffW1T[ffi], ff[ffi].b1, nullptr, Z, 128, 256);
            ln_gelu_k<<<Nn, 256, 0, stream>>>(Z, ff[ffi].lng, ff[ffi].lnb, 256);
            cvt16_k<<<g1(NF2), 256, 0, stream>>>(Z, Z16, NF2);
            gemm_wmma<2><<<MBLK, 256, 0, stream>>>(Z16, ffW2T[ffi], ff[ffi].b2, ls[li], H, 256, 128);
            cvt16_k<<<g1(NF), 256, 0, stream>>>(H, H16, NF);   // for next conv
        }
    };
    run_layer(0, true, 0);
    run_layer(1, true, 1);
    run_layer(2, false, 0);

    // ---- pooling (mean | max | sum) per graph
    fill_f32_k<<<g1(NGRAPH * 128), 256, 0, stream>>>(psum, 0.f, NGRAPH * 128);
    fill_u32_k<<<g1(NGRAPH * 128), 256, 0, stream>>>(pmax, ENC_NEG_INF, NGRAPH * 128);
    fill_f32_k<<<g1(NGRAPH), 256, 0, stream>>>(cnt, 0.f, NGRAPH);
    pool_accum_k<<<g1(NF), 256, 0, stream>>>(H, batch, psum, pmax, cnt, Nn);
    pool_final_k<<<g1(NGRAPH * 128), 256, 0, stream>>>(psum, pmax, cnt, pooled);
    cvt16_k<<<g1(NGRAPH * 384), 256, 0, stream>>>(pooled, pooled16, NGRAPH * 384);

    // ---- classification heads
    for (int i = 0; i < 2; ++i) {
        gemm_wmma<0><<<NGRAPH / 16, 256, 0, stream>>>(pooled16, hdW1T[i], hd[i].b1,
                                                      nullptr, ZH, 384, 128);
        ln_gelu_k<<<NGRAPH, 128, 0, stream>>>(ZH, hd[i].lng, hd[i].lnb, 128);
        head_out_k<<<g1((long long)NGRAPH * ncs[i]), 256, 0, stream>>>(
            ZH, hd[i].w2, hd[i].b2, out_ptr[i], ncs[i]);
    }
}
